// SlidingWindowAttention_33517924778610
// MI455X (gfx1250) — matrix operand order repaired and checked by reading, not compiled, not run on hardware
//
#include <hip/hip_runtime.h>
#include <stdint.h>

typedef __attribute__((ext_vector_type(16))) __bf16 v16bf;
typedef __attribute__((ext_vector_type(8)))  float  v8f;

union FragB16 {
    v16bf v;
    unsigned short u[16];
    uint4 q[2];
};

__device__ __forceinline__ unsigned short f2bf(float f) {
    union { float f; uint32_t u; } c;
    c.f = f;
    uint32_t u = c.u;
    uint32_t r = u + 0x7FFFu + ((u >> 16) & 1u);  // round-to-nearest-even
    return (unsigned short)(r >> 16);
}

// ---------------------------------------------------------------------------
// f32 -> bf16 conversion, 4 elements per thread (bandwidth-trivial)
// ---------------------------------------------------------------------------
__global__ void cvt_f32_bf16(const float* __restrict__ in,
                             unsigned short* __restrict__ out, int n) {
    int i = (blockIdx.x * blockDim.x + threadIdx.x) * 4;
    if (i + 3 < n) {
        float4 v = *(const float4*)(in + i);
        uint2 o;
        o.x = (unsigned)f2bf(v.x) | ((unsigned)f2bf(v.y) << 16);
        o.y = (unsigned)f2bf(v.z) | ((unsigned)f2bf(v.w) << 16);
        *(uint2*)(out + i) = o;
    }
}

// ---------------------------------------------------------------------------
// C[M,N] = A[M,K] * W[N,K]^T   (both operands K-contiguous; bf16 in, f32 acc)
// Block: 256 threads (8 waves). Block tile 128x128, BK=32.
// Waves arranged 2(m) x 4(n); each wave computes 64x32 = 8 WMMA tiles.
// Fragment layouts used here (wave32, contiguous k for both A and B):
//   A 16x32 bf16: lane<16 holds K 0-15, lane>=16 holds K 16-31; row = lane&15
//   B 32x16 bf16: lane<16 holds K 0-15, lane>=16 holds K 16-31; col = lane&15
// ---------------------------------------------------------------------------
template<bool OUT_BF16>
__global__ __launch_bounds__(256) void gemm_xwt(
    const unsigned short* __restrict__ A,   // [M][K]
    const unsigned short* __restrict__ W,   // [N][K]
    void* __restrict__ Cout, int M, int N, int K)
{
    __shared__ unsigned short lA[128 * 32];
    __shared__ unsigned short lW[128 * 32];

    const int tid  = threadIdx.x;
    const int lane = tid & 31, wave = tid >> 5;
    const int wm = wave >> 2;      // 0..1 -> 64 rows each
    const int wn = wave & 3;       // 0..3 -> 32 cols each
    const int rowBase = blockIdx.y * 128;
    const int colBase = blockIdx.x * 128;
    const int l    = lane & 15;
    const int khA  = (lane < 16) ? 0 : 16;   // A-fragment K pack offset
    const int kh16 = (lane < 16) ? 0 : 16;   // B-fragment K half offset

    v8f acc[4][2];
    for (int s = 0; s < 4; s++)
        for (int t = 0; t < 2; t++)
            for (int r = 0; r < 8; r++) acc[s][t][r] = 0.0f;

    for (int k0 = 0; k0 < K; k0 += 32) {
        __syncthreads();
        // Stage 128x32 tiles of A and W: 512 16B chunks each, 2 per thread
        #pragma unroll
        for (int i = 0; i < 2; i++) {
            int c = tid + i * 256;
            int row = c >> 2, seg = c & 3;
            ((uint4*)lA)[c] = *(const uint4*)(A + (size_t)(rowBase + row) * K + k0 + seg * 8);
            ((uint4*)lW)[c] = *(const uint4*)(W + (size_t)(colBase + row) * K + k0 + seg * 8);
        }
        __syncthreads();

        FragB16 fa[4], fb[2];
        #pragma unroll
        for (int s = 0; s < 4; s++) {
            const unsigned short* p = &lA[(wm * 64 + s * 16 + l) * 32 + khA];
            fa[s].q[0] = *(const uint4*)p;
            fa[s].q[1] = *(const uint4*)(p + 8);
        }
        #pragma unroll
        for (int t = 0; t < 2; t++) {
            const unsigned short* p = &lW[(wn * 32 + t * 16 + l) * 32 + kh16];
            fb[t].q[0] = *(const uint4*)p;
            fb[t].q[1] = *(const uint4*)(p + 8);
        }
        #pragma unroll
        for (int s = 0; s < 4; s++)
            #pragma unroll
            for (int t = 0; t < 2; t++)
                acc[s][t] = __builtin_amdgcn_wmma_f32_16x16x32_bf16(
                    false, fa[s].v, false, fb[t].v, (short)0, acc[s][t], false, false);
    }

    // Store: C layout -> lane<16 rows r, lane>=16 rows r+8; col = lane&15
    const int mOff = (lane < 16) ? 0 : 8;
    for (int s = 0; s < 4; s++)
        for (int t = 0; t < 2; t++) {
            int n = colBase + wn * 32 + t * 16 + l;
            #pragma unroll
            for (int r = 0; r < 8; r++) {
                int m = rowBase + wm * 64 + s * 16 + mOff + r;
                float val = acc[s][t][r];
                if constexpr (OUT_BF16)
                    ((unsigned short*)Cout)[(size_t)m * N + n] = f2bf(val);
                else
                    ((float*)Cout)[(size_t)m * N + n] = val;
            }
        }
}

// ---------------------------------------------------------------------------
// Sliding-window flash attention. Grid: (S/128, H). Block: 256 (8 waves).
// The block owns 128 q rows of one head; all waves iterate the UNION key
// window in 32-key blocks, cooperatively staging K and V tiles in LDS
// (coalesced b128 loads, shared 8 ways). Each wave computes its own 16-row
// S tile (4 WMMAs over D=128 per 16-key subtile), does online softmax with
// shfl_xor row reductions (stay inside 16-lane halves), transposes P via
// per-wave LDS scratch into A-fragment layout, then PV: 8 WMMAs with the
// V B-fragment gathered from the padded LDS tile (bank-conflict-free
// column reads, stride 136 halves -> 68 dwords -> bank step 4).
// ---------------------------------------------------------------------------
__global__ __launch_bounds__(256) void attn_swa(
    const unsigned short* __restrict__ Q,
    const unsigned short* __restrict__ Km,
    const unsigned short* __restrict__ V,
    unsigned short* __restrict__ O)
{
    constexpr int E = 2048, D = 128, WIN = 1024;
    constexpr int KP = 136;                      // padded LDS row stride (halves)
    __shared__ unsigned short Kt[32 * KP];       // 8704 B
    __shared__ unsigned short Vt[32 * KP];       // 8704 B
    __shared__ unsigned short Pl[8][16 * 32];    // 8192 B

    const int tid  = threadIdx.x;
    const int lane = tid & 31, wave = tid >> 5;
    const int l     = lane & 15;
    const int khA   = (lane < 16) ? 0 : 8;
    const int kh16  = (lane < 16) ? 0 : 16;
    const int mOff  = (lane < 16) ? 0 : 8;
    const int qblk  = blockIdx.x * 128;
    const int q0    = qblk + wave * 16;
    const int hbase = blockIdx.y * D;
    const float scale = 0.08838834764831845f;    // 1/sqrt(128)
    const float NEG = -3.0e38f;

    // Q fragments (A operand), 4 chunks over D, loaded once from global
    FragB16 qf[4];
    {
        const unsigned short* p = Q + (size_t)(q0 + l) * E + hbase + kh16;
        #pragma unroll
        for (int j = 0; j < 4; j++) {
            qf[j].q[0] = *(const uint4*)(p + j * 32);
            qf[j].q[1] = *(const uint4*)(p + j * 32 + 8);
        }
    }

    v8f oacc[8];
    float mrun[8], lrun[8];
    for (int d = 0; d < 8; d++)
        for (int r = 0; r < 8; r++) oacc[d][r] = 0.0f;
    for (int r = 0; r < 8; r++) { mrun[r] = NEG; lrun[r] = 0.0f; }

    // Union window for the whole 128-row block
    int kstart = qblk - (WIN - 1);
    if (kstart < 0) kstart = 0;
    kstart &= ~31;
    const int kend = qblk + 127;

    for (int kb = kstart; kb <= kend; kb += 32) {
        __syncthreads();
        // Cooperative staging: 32 keys x 128 d, K and V, 512 16B chunks each
        #pragma unroll
        for (int i = 0; i < 2; i++) {
            int c = tid + i * 256;
            int key = c >> 4, seg = c & 15;
            size_t goff = (size_t)(kb + key) * E + hbase + seg * 8;
            *(uint4*)&Kt[key * KP + seg * 8] = *(const uint4*)(Km + goff);
            *(uint4*)&Vt[key * KP + seg * 8] = *(const uint4*)(V + goff);
        }
        __syncthreads();

        // Per-wave uniform skip (both barriers above already executed)
        if (kb > q0 + 15 || kb + 31 < q0 - (WIN - 1)) continue;

        v8f s0, s1;
        for (int r = 0; r < 8; r++) { s0[r] = 0.0f; s1[r] = 0.0f; }

        // S = Q * K^T from LDS (two 16-key subtiles, 4 d-chunks each)
        #pragma unroll
        for (int j = 0; j < 4; j++) {
            FragB16 kf0, kf1;
            const unsigned short* p0 = &Kt[l * KP        + j * 32 + kh16];
            const unsigned short* p1 = &Kt[(16 + l) * KP + j * 32 + kh16];
            kf0.q[0] = *(const uint4*)p0; kf0.q[1] = *(const uint4*)(p0 + 8);
            kf1.q[0] = *(const uint4*)p1; kf1.q[1] = *(const uint4*)(p1 + 8);
            s0 = __builtin_amdgcn_wmma_f32_16x16x32_bf16(false, qf[j].v, false, kf0.v, (short)0, s0, false, false);
            s1 = __builtin_amdgcn_wmma_f32_16x16x32_bf16(false, qf[j].v, false, kf1.v, (short)0, s1, false, false);
        }

        // Mask + online softmax; write P (bf16) into per-wave LDS region
        #pragma unroll
        for (int r = 0; r < 8; r++) {
            int qrow = q0 + mOff + r;
            int k0i = kb + l, k1i = k0i + 16;
            float a = s0[r] * scale, b = s1[r] * scale;
            if (k0i > qrow || k0i <= qrow - WIN) a = NEG;
            if (k1i > qrow || k1i <= qrow - WIN) b = NEG;
            float mx = fmaxf(a, b);
            for (int off = 1; off < 16; off <<= 1) mx = fmaxf(mx, __shfl_xor(mx, off, 32));
            float mnew = fmaxf(mrun[r], mx);
            float corr = __expf(mrun[r] - mnew);
            float pa = (a == NEG) ? 0.0f : __expf(a - mnew);
            float pb = (b == NEG) ? 0.0f : __expf(b - mnew);
            float rs = pa + pb;
            for (int off = 1; off < 16; off <<= 1) rs += __shfl_xor(rs, off, 32);
            lrun[r] = lrun[r] * corr + rs;
            mrun[r] = mnew;
            #pragma unroll
            for (int d = 0; d < 8; d++) oacc[d][r] *= corr;
            Pl[wave][(mOff + r) * 32 + l]      = f2bf(pa);
            Pl[wave][(mOff + r) * 32 + 16 + l] = f2bf(pb);
        }

        // Re-read P as an A-fragment (16x32) — same-wave LDS ops are in-order
        FragB16 pf;
        {
            const unsigned short* pp = &Pl[wave][l * 32 + kh16];
            pf.q[0] = *(const uint4*)pp;
            pf.q[1] = *(const uint4*)(pp + 8);
        }

        // O += P * V : V B-fragment = column read from padded LDS tile
        #pragma unroll
        for (int d = 0; d < 8; d++) {
            FragB16 vf;
            const unsigned short* pv = &Vt[kh16 * KP + d * 16 + l];
            #pragma unroll
            for (int i = 0; i < 16; i++) vf.u[i] = pv[i * KP];
            oacc[d] = __builtin_amdgcn_wmma_f32_16x16x32_bf16(
                false, pf.v, false, vf.v, (short)0, oacc[d], false, false);
        }
    }

    // Finalize: divide by row sums, store bf16
    #pragma unroll
    for (int r = 0; r < 8; r++) {
        float inv = 1.0f / lrun[r];
        int qrow = q0 + mOff + r;
        #pragma unroll
        for (int d = 0; d < 8; d++)
            O[(size_t)qrow * E + hbase + d * 16 + l] = f2bf(oacc[d][r] * inv);
    }
}

// ---------------------------------------------------------------------------
// Host-side pipeline
// ---------------------------------------------------------------------------
extern "C" void kernel_launch(void* const* d_in, const int* in_sizes, int n_in,
                              void* d_out, int out_size, void* d_ws, size_t ws_size,
                              hipStream_t stream)
{
    const int S = 4096, E = 2048;
    const float* x  = (const float*)d_in[0];
    const float* Wq = (const float*)d_in[1];
    const float* Wk = (const float*)d_in[2];
    const float* Wv = (const float*)d_in[3];
    const float* Wo = (const float*)d_in[4];

    const size_t nx = (size_t)S * E;   // 8,388,608
    const size_t nw = (size_t)E * E;   // 4,194,304

    unsigned short* ws  = (unsigned short*)d_ws;
    unsigned short* xb  = ws;
    unsigned short* wqb = xb  + nx;
    unsigned short* wkb = wqb + nw;
    unsigned short* wvb = wkb + nw;
    unsigned short* wob = wvb + nw;
    unsigned short* Qb  = wob + nw;
    unsigned short* Kb  = Qb  + nx;
    unsigned short* Vb  = Kb  + nx;
    unsigned short* Ab  = Vb  + nx;

    // 1) f32 -> bf16 conversions
    cvt_f32_bf16<<<dim3((unsigned)((nx / 4 + 255) / 256)), 256, 0, stream>>>(x,  xb,  (int)nx);
    cvt_f32_bf16<<<dim3((unsigned)((nw / 4 + 255) / 256)), 256, 0, stream>>>(Wq, wqb, (int)nw);
    cvt_f32_bf16<<<dim3((unsigned)((nw / 4 + 255) / 256)), 256, 0, stream>>>(Wk, wkb, (int)nw);
    cvt_f32_bf16<<<dim3((unsigned)((nw / 4 + 255) / 256)), 256, 0, stream>>>(Wv, wvb, (int)nw);
    cvt_f32_bf16<<<dim3((unsigned)((nw / 4 + 255) / 256)), 256, 0, stream>>>(Wo, wob, (int)nw);

    // 2) Q/K/V projections (bf16 out)
    dim3 gg(E / 128, S / 128);
    gemm_xwt<true><<<gg, 256, 0, stream>>>(xb, wqb, Qb, S, E, E);
    gemm_xwt<true><<<gg, 256, 0, stream>>>(xb, wkb, Kb, S, E, E);
    gemm_xwt<true><<<gg, 256, 0, stream>>>(xb, wvb, Vb, S, E, E);

    // 3) Sliding-window attention
    attn_swa<<<dim3(S / 128, 16), 256, 0, stream>>>(Qb, Kb, Vb, Ab);

    // 4) Output projection (f32 out)
    gemm_xwt<false><<<gg, 256, 0, stream>>>(Ab, wob, d_out, S, E, E);
}
